// TextEncoder_35399120454088
// MI455X (gfx1250) — compile-verified, matrix operand order checked by reading, not run on hardware
//
#include <hip/hip_runtime.h>
#include <hip/hip_bf16.h>

// ---- CDNA5 (gfx1250) fused LSTM text encoder ------------------------------
// emb lookup + input projection + 64-step recurrence fused in one persistent
// kernel; all weights resident in LDS as bf16; GEMMs via v_wmma_f32_16x16x32_bf16.
// v2: bias fed via WMMA C operand (no acc-init movs), W_hh fragments hoisted
// to registers, double-buffered embedding staging to hide gather latency.

typedef __attribute__((ext_vector_type(16))) __bf16 v16bf;
typedef __attribute__((ext_vector_type(8)))  float  v8f;
typedef __attribute__((ext_vector_type(4)))  int    i32x4;
typedef __attribute__((ext_vector_type(4)))  float  f32x4;

#define VOCABN  32000
#define EMBEDN  256
#define HIDDENN 64
#define NGATES  256      // 4*HIDDEN
#define BATCHN  4096
#define TSTEPS  64
#define ROWS    32       // batch rows per block
#define THREADS 256      // 8 wave32s

// padded LDS strides (elements) to avoid bank conflicts on b128 fragment loads
#define WIH_S 264   // bf16
#define WHH_S 72    // bf16
#define EMB_S 264   // bf16
#define H_S   72    // bf16
#define G_S   260   // f32

// LDS layout (bytes, all 16B aligned)
#define OFF_WIH  0
#define OFF_WHH  135168                    // 256*264*2
#define OFF_EMB0 (OFF_WHH + 36864)         // 256*72*2
#define OFF_EMB1 (OFF_EMB0 + 16896)        // 32*264*2
#define OFF_H    (OFF_EMB1 + 16896)        // 32*264*2
#define OFF_G    (OFF_H   + 4608)          // 32*72*2
#define OFF_BIAS (OFF_G   + 33280)         // 32*260*4
#define OFF_IDX  (OFF_BIAS + 1024)         // 256*4
#define SMEM_BYTES (OFF_IDX + 8192)        // + 32*64*4 indices = 252928 B

union FragBF { v16bf v; i32x4 q[2]; };

__device__ __forceinline__ float fast_sigmoid(float x) {
  return 1.f / (1.f + __expf(-x));
}
__device__ __forceinline__ float fast_tanh(float x) {
  return 1.f - 2.f / (1.f + __expf(2.f * x));
}

__global__ __launch_bounds__(THREADS)
void lstm_fused_kernel(const int*   __restrict__ x,
                       const float* __restrict__ embed_table,
                       const float* __restrict__ W_ih,
                       const float* __restrict__ W_hh,
                       const float* __restrict__ b_ih,
                       const float* __restrict__ b_hh,
                       float*       __restrict__ out)
{
  extern __shared__ __align__(16) char smem[];
  __bf16* wih   = (__bf16*)(smem + OFF_WIH);   // [NGATES][WIH_S]  (row = gate n, col = k)
  __bf16* whh   = (__bf16*)(smem + OFF_WHH);   // [NGATES][WHH_S]
  __bf16* embA  = (__bf16*)(smem + OFF_EMB0);  // [ROWS][EMB_S]  ping
  __bf16* embB  = (__bf16*)(smem + OFF_EMB1);  // [ROWS][EMB_S]  pong
  __bf16* hbuf  = (__bf16*)(smem + OFF_H);     // [ROWS][H_S]
  float*  gbuf  = (float* )(smem + OFF_G);     // [ROWS][G_S]
  float*  bias  = (float* )(smem + OFF_BIAS);  // [NGATES]
  int*    idxb  = (int*  )(smem + OFF_IDX);    // [ROWS][TSTEPS]

  const int tid  = threadIdx.x;
  const int lane = tid & 31;
  const int wv   = tid >> 5;
  const int b0   = blockIdx.x * ROWS;

  const int em   = tid >> 3;         // embedding stage: row (0..31)
  const int eseg = (tid & 7) * 32;   // 32-col segment

  // ---------------- prologue: stage weights (fp32 -> bf16 LDS) -------------
  {
    const f32x4* s = (const f32x4*)(W_ih + tid * EMBEDN);
    __bf16* d = wih + tid * WIH_S;
    #pragma unroll 8
    for (int c = 0; c < EMBEDN / 4; ++c) {
      f32x4 f = s[c];
      d[c*4+0] = (__bf16)f[0]; d[c*4+1] = (__bf16)f[1];
      d[c*4+2] = (__bf16)f[2]; d[c*4+3] = (__bf16)f[3];
    }
    const f32x4* s2 = (const f32x4*)(W_hh + tid * HIDDENN);
    __bf16* d2 = whh + tid * WHH_S;
    #pragma unroll
    for (int c = 0; c < HIDDENN / 4; ++c) {
      f32x4 f = s2[c];
      d2[c*4+0] = (__bf16)f[0]; d2[c*4+1] = (__bf16)f[1];
      d2[c*4+2] = (__bf16)f[2]; d2[c*4+3] = (__bf16)f[3];
    }
    bias[tid] = b_ih[tid] + b_hh[tid];
    for (int i = tid; i < ROWS * H_S; i += THREADS) hbuf[i] = (__bf16)0.f;
    for (int i = tid; i < ROWS * TSTEPS; i += THREADS) idxb[i] = x[b0 * TSTEPS + i];
  }
  __syncthreads();

  // ---- stage embeddings for t = 0 into ping buffer ----
  {
    const size_t vidx = (size_t)idxb[em * TSTEPS + 0];
    const f32x4* s = (const f32x4*)(embed_table + vidx * EMBEDN + eseg);
    __bf16* d = embA + em * EMB_S + eseg;
    #pragma unroll
    for (int c = 0; c < 8; ++c) {
      f32x4 f = s[c];
      d[c*4+0] = (__bf16)f[0]; d[c*4+1] = (__bf16)f[1];
      d[c*4+2] = (__bf16)f[2]; d[c*4+3] = (__bf16)f[3];
    }
  }

  // per-wave WMMA tile coordinates: 2 M-tiles x 16 N-tiles over 8 waves
  const int m_tile = wv & 1;                    // which 16 batch rows
  const int n0     = (wv >> 1) * 64;            // 64-wide gate column span
  const int arow   = m_tile * 16 + (lane & 15); // A-fragment row
  const int koff   = (lane >> 4) << 3;          // K half-select (0 or 8)

  // persistent bias broadcast in C/D layout (lane = n, all 8 vgprs same val)
  v8f bias_vec[4];
  #pragma unroll
  for (int j = 0; j < 4; ++j) {
    const float bv = bias[n0 + j * 16 + (lane & 15)];
    v8f a;
    #pragma unroll
    for (int v = 0; v < 8; ++v) a[v] = bv;
    bias_vec[j] = a;
  }

  // persistent W_hh B-fragments (loop invariant): 2 K-tiles x 4 N-tiles
  FragBF whf[2][4];
  #pragma unroll
  for (int k = 0; k < 2; ++k)
    #pragma unroll
    for (int j = 0; j < 4; ++j) {
      const __bf16* bp = whh + (n0 + j * 16 + (lane & 15)) * WHH_S + k * 32 + koff;
      whf[k][j].q[0] = *(const i32x4*)bp;
      whf[k][j].q[1] = *(const i32x4*)(bp + 16);
    }

  float c_reg[8];
  #pragma unroll
  for (int k = 0; k < 8; ++k) c_reg[k] = 0.f;

  __syncthreads();

  for (int t = 0; t < TSTEPS; ++t) {
    __bf16* cur = (t & 1) ? embB : embA;
    __bf16* nxt = (t & 1) ? embA : embB;

    // ---- stage embeddings for step t+1 into the other buffer (overlaps
    //      with this step's WMMAs; consumed only after next barrier pair) ----
    if (t + 1 < TSTEPS) {
      const size_t vidx = (size_t)idxb[em * TSTEPS + t + 1];
      const f32x4* s = (const f32x4*)(embed_table + vidx * EMBEDN + eseg);
      __bf16* d = nxt + em * EMB_S + eseg;
      #pragma unroll
      for (int c = 0; c < 8; ++c) {
        f32x4 f = s[c];
        d[c*4+0] = (__bf16)f[0]; d[c*4+1] = (__bf16)f[1];
        d[c*4+2] = (__bf16)f[2]; d[c*4+3] = (__bf16)f[3];
      }
      if (t + 2 < TSTEPS && (tid & 7) == 0) {  // warm L2 two steps ahead
        const size_t nidx = (size_t)idxb[em * TSTEPS + t + 2];
        __builtin_prefetch(embed_table + nidx * EMBEDN, 0, 0);
      }
    }

    // ---- gates = emb @ Wih^T + h @ Whh^T + bias  (all WMMA bf16) ----
    v8f acc[4];

    // K-tile 0: bias enters through the C operand (no accumulator-init movs)
    {
      FragBF a;
      const __bf16* ap = cur + arow * EMB_S + koff;
      a.q[0] = *(const i32x4*)ap;
      a.q[1] = *(const i32x4*)(ap + 16);
      #pragma unroll
      for (int j = 0; j < 4; ++j) {
        FragBF b;
        const __bf16* bp = wih + (n0 + j * 16 + (lane & 15)) * WIH_S + koff;
        b.q[0] = *(const i32x4*)bp;
        b.q[1] = *(const i32x4*)(bp + 16);
        acc[j] = __builtin_amdgcn_wmma_f32_16x16x32_bf16(
            false, a.v, false, b.v, (short)0, bias_vec[j], false, false);
      }
    }
    // K-tiles 1..7 of the input projection
    #pragma unroll
    for (int k = 1; k < 8; ++k) {
      FragBF a;
      const __bf16* ap = cur + arow * EMB_S + k * 32 + koff;
      a.q[0] = *(const i32x4*)ap;
      a.q[1] = *(const i32x4*)(ap + 16);
      #pragma unroll
      for (int j = 0; j < 4; ++j) {
        FragBF b;
        const __bf16* bp = wih + (n0 + j * 16 + (lane & 15)) * WIH_S + k * 32 + koff;
        b.q[0] = *(const i32x4*)bp;
        b.q[1] = *(const i32x4*)(bp + 16);
        acc[j] = __builtin_amdgcn_wmma_f32_16x16x32_bf16(
            false, a.v, false, b.v, (short)0, acc[j], false, false);
      }
    }
    // recurrence (K = 64), W_hh fragments already in registers
    #pragma unroll
    for (int k = 0; k < 2; ++k) {
      FragBF a;
      const __bf16* ap = hbuf + arow * H_S + k * 32 + koff;
      a.q[0] = *(const i32x4*)ap;
      a.q[1] = *(const i32x4*)(ap + 16);
      #pragma unroll
      for (int j = 0; j < 4; ++j) {
        acc[j] = __builtin_amdgcn_wmma_f32_16x16x32_bf16(
            false, a.v, false, whf[k][j].v, (short)0, acc[j], false, false);
      }
    }

    // ---- scatter gate pre-activations to LDS (C-layout: lane=n, vgpr=m) ----
    {
      const int mb = m_tile * 16 + ((lane >> 4) << 3);
      #pragma unroll
      for (int j = 0; j < 4; ++j) {
        const int n = n0 + j * 16 + (lane & 15);
        #pragma unroll
        for (int v = 0; v < 8; ++v)
          gbuf[(mb + v) * G_S + n] = acc[j][v];
      }
    }
    __syncthreads();

    // ---- elementwise LSTM cell: 2048 states, 8 per thread, c in registers --
    #pragma unroll
    for (int k = 0; k < 8; ++k) {
      const int e = tid + k * THREADS;
      const int m = e >> 6;
      const int h = e & 63;
      const float gi = gbuf[m * G_S +       h];
      const float gf = gbuf[m * G_S +  64 + h];
      const float gg = gbuf[m * G_S + 128 + h];
      const float go = gbuf[m * G_S + 192 + h];
      const float c  = fast_sigmoid(gf) * c_reg[k] + fast_sigmoid(gi) * fast_tanh(gg);
      c_reg[k] = c;
      const float hv = fast_sigmoid(go) * fast_tanh(c);
      hbuf[m * H_S + h] = (__bf16)hv;
      if (t == TSTEPS - 1)
        out[(size_t)(b0 + m) * HIDDENN + h] = hv;
    }
    __syncthreads();
  }
}

extern "C" void kernel_launch(void* const* d_in, const int* in_sizes, int n_in,
                              void* d_out, int out_size, void* d_ws, size_t ws_size,
                              hipStream_t stream) {
  (void)in_sizes; (void)n_in; (void)out_size; (void)d_ws; (void)ws_size;
  const int*   x           = (const int*)  d_in[0];
  const float* embed_table = (const float*)d_in[1];
  const float* W_ih        = (const float*)d_in[2];
  const float* W_hh        = (const float*)d_in[3];
  const float* b_ih        = (const float*)d_in[4];
  const float* b_hh        = (const float*)d_in[5];
  float* out = (float*)d_out;

  // allow >64KB dynamic LDS (WGP supports up to 320KB on gfx1250)
  hipFuncSetAttribute(reinterpret_cast<const void*>(lstm_fused_kernel),
                      hipFuncAttributeMaxDynamicSharedMemorySize, SMEM_BYTES);

  lstm_fused_kernel<<<dim3(BATCHN / ROWS), dim3(THREADS), SMEM_BYTES, stream>>>(
      x, embed_table, W_ih, W_hh, b_ih, b_hh, out);
}
